// AzureMLGraphAttentionNetwork_55018531062293
// MI455X (gfx1250) — matrix operand order repaired and checked by reading, not compiled, not run on hardware
//
#include <hip/hip_runtime.h>
#include <hip/hip_bf16.h>

// ---------------------------------------------------------------------------
// GAT (3 layers) for MI455X / gfx1250.
// GEMMs: bf16 WMMA (v_wmma_f32_16x16x32_bf16), f32 accumulate,
//        16x(16*G) output tile per wave: 1 A-fragment feeds G WMMAs per k-step.
// Edge softmax/aggregation: L2-resident gathers + hardware f32 atomics.
// ---------------------------------------------------------------------------

typedef __attribute__((ext_vector_type(16))) __bf16 v16bf;
typedef __attribute__((ext_vector_type(8)))  __bf16 v8bf;
typedef __attribute__((ext_vector_type(8)))  float  v8f;

// ---------------- small utility kernels ----------------

__global__ void k_cast_bf16(const float* __restrict__ x, __bf16* __restrict__ y, long n) {
    long i = (long)blockIdx.x * blockDim.x + threadIdx.x;
    if (i < n) y[i] = (__bf16)x[i];
}

__global__ void k_fill_f32(float* __restrict__ p, float v, long n) {
    long i = (long)blockIdx.x * blockDim.x + threadIdx.x;
    if (i < n) p[i] = v;
}

// W [K x M] f32 row-major  ->  Wt [M x K] bf16 row-major (B operand, K contiguous)
__global__ void k_transpose_bf16(const float* __restrict__ W, __bf16* __restrict__ Wt,
                                 int K, int M) {
    int i = blockIdx.x * blockDim.x + threadIdx.x;
    if (i >= K * M) return;
    int k = i / M, m = i % M;
    Wt[(long)m * K + k] = (__bf16)W[i];
}

// ---------------- WMMA GEMM: C[N,M] = A[N,K] x Wt[M,K]^T ----------------
// One wave per 16 x (16*G) output tile; K-loop in steps of 32.
// A frag (16x32 bf16): lane -> row=lane&15, elems j -> K=(lane>>4)*16+j.
//   B frag (32x16): lane -> col=lane&15, elems j -> K=(lane>>4)*16+j.
// C/D: lane -> col=lane&15, vgpr r -> row = r + 8*(lane>>4).
// The single A fragment per k-step is reused by all G WMMAs (4x less A traffic).
template <int G>
__global__ void k_wmma_gemm_bf16(const __bf16* __restrict__ A,
                                 const __bf16* __restrict__ Bt,
                                 float* __restrict__ C,
                                 int Nrows, int K, int M) {
    int wid  = (blockIdx.x * blockDim.x + threadIdx.x) >> 5;
    int lane = threadIdx.x & 31;
    int groups_n = M / (16 * G);
    int total    = (Nrows >> 4) * groups_n;
    if (wid >= total) return;
    int tm  = wid / groups_n;
    int tg  = wid % groups_n;
    int half = lane >> 4;    // 0 or 1
    int l15  = lane & 15;

    const __bf16* Arow = A + (long)(tm * 16 + l15) * K + half * 16;
    const __bf16* Brow[G];
#pragma unroll
    for (int j = 0; j < G; ++j)
        Brow[j] = Bt + (long)(tg * 16 * G + j * 16 + l15) * K + half * 16;

    v8f acc[G];
#pragma unroll
    for (int j = 0; j < G; ++j) acc[j] = (v8f){};

    for (int k0 = 0; k0 < K; k0 += 32) {
        v8bf alo = *(const v8bf*)(Arow + k0);        // K = k0+16*half .. +7
        v8bf ahi = *(const v8bf*)(Arow + k0 + 8);    // K = k0+16*half+8 .. +7
        v16bf a = __builtin_shufflevector(alo, ahi,
                     0,1,2,3,4,5,6,7,8,9,10,11,12,13,14,15);
#pragma unroll
        for (int j = 0; j < G; ++j) {
            v16bf b = *(const v16bf*)(Brow[j] + k0); // 16 contiguous K values
            acc[j] = __builtin_amdgcn_wmma_f32_16x16x32_bf16(
                         false, a, false, b, (short)0, acc[j], false, false);
        }
    }

#pragma unroll
    for (int j = 0; j < G; ++j) {
        float* Crow = C + (long)(tm * 16 + half * 8) * M + tg * 16 * G + j * 16 + l15;
#pragma unroll
        for (int r = 0; r < 8; ++r) Crow[(long)r * M] = acc[j][r];
    }
}

// ---------------- attention scores + segment-softmax state init ----------------

__global__ void k_scores_init(const float* __restrict__ h,
                              const float* __restrict__ a_s,
                              const float* __restrict__ a_d,
                              float* __restrict__ e_src, float* __restrict__ e_dst,
                              float* __restrict__ emax,  float* __restrict__ denom,
                              int N, int H, int F, int M) {
    int i = blockIdx.x * blockDim.x + threadIdx.x;
    if (i >= N * H) return;
    int n = i / H, hh = i % H;
    const float* hp = h + (long)n * M + hh * F;
    const float* as = a_s + hh * F;
    const float* ad = a_d + hh * F;
    float ss = 0.f, sd = 0.f;
    for (int f = 0; f < F; ++f) { float hv = hp[f]; ss += hv * as[f]; sd += hv * ad[f]; }
    e_src[i] = ss;
    e_dst[i] = sd;
    emax[i]  = -INFINITY;
    denom[i] = 0.f;
}

// ---------------- edge passes ----------------

__device__ inline float leaky(float e) { return e > 0.f ? e : 0.2f * e; }

__device__ inline void atomicMaxF(float* addr, float v) {
    // float max via signed/unsigned integer ordering trick (no CAS loop)
    if (v >= 0.f) atomicMax((int*)addr, __float_as_int(v));
    else          atomicMin((unsigned int*)addr, __float_as_uint(v));
}

__device__ inline void edge_sd(const int* esrc, const int* edst, int E, int e,
                               int& s, int& d) {
    if (e < E) { s = esrc[e]; d = edst[e]; }
    else       { s = e - E;  d = s; }       // self-loop
}

__global__ void k_edge_max(const int* __restrict__ esrc, const int* __restrict__ edst,
                           int E, int Etot,
                           const float* __restrict__ e_src, const float* __restrict__ e_dst,
                           float* __restrict__ emax, int H) {
    int i = blockIdx.x * blockDim.x + threadIdx.x;
    if (i >= Etot * H) return;
    int e = i / H, hh = i % H, s, d;
    edge_sd(esrc, edst, E, e, s, d);
    float ev = leaky(e_src[s * H + hh] + e_dst[d * H + hh]);
    atomicMaxF(&emax[d * H + hh], ev);
}

__global__ void k_edge_expsum(const int* __restrict__ esrc, const int* __restrict__ edst,
                              int E, int Etot,
                              const float* __restrict__ e_src, const float* __restrict__ e_dst,
                              const float* __restrict__ emax, float* __restrict__ denom, int H) {
    int i = blockIdx.x * blockDim.x + threadIdx.x;
    if (i >= Etot * H) return;
    int e = i / H, hh = i % H, s, d;
    edge_sd(esrc, edst, E, e, s, d);
    float ev = leaky(e_src[s * H + hh] + e_dst[d * H + hh]);
    float ex = __expf(ev - emax[d * H + hh]);
    unsafeAtomicAdd(&denom[d * H + hh], ex);
}

// one block per edge: alpha per head in LDS, then scatter-add h[src]*alpha into acc[dst]
__global__ void k_edge_aggregate(const int* __restrict__ esrc, const int* __restrict__ edst,
                                 int E, int Etot,
                                 const float* __restrict__ h,
                                 const float* __restrict__ e_src, const float* __restrict__ e_dst,
                                 const float* __restrict__ emax,  const float* __restrict__ denom,
                                 float* __restrict__ acc,
                                 int H, int logF, int M) {
    int e = blockIdx.x;
    if (e >= Etot) return;
    int s, d;
    edge_sd(esrc, edst, E, e, s, d);

    __shared__ float alpha[8];
    if ((int)threadIdx.x < H) {
        int hh = threadIdx.x;
        float ev = leaky(e_src[s * H + hh] + e_dst[d * H + hh]);
        float ex = __expf(ev - emax[d * H + hh]);
        alpha[hh] = ex / (denom[d * H + hh] + 1e-16f);
    }
    __syncthreads();

    const float* hs = h   + (long)s * M;
    float*       ac = acc + (long)d * M;
    for (int f = threadIdx.x; f < M; f += blockDim.x)
        unsafeAtomicAdd(&ac[f], hs[f] * alpha[f >> logF]);
}

// ---------------- epilogue: +bias, optional ReLU, f32 and/or bf16 out ----------------

__global__ void k_epilogue(const float* __restrict__ acc, const float* __restrict__ bias,
                           float* __restrict__ outf, __bf16* __restrict__ outb,
                           long NM, int M, int relu) {
    long i = (long)blockIdx.x * blockDim.x + threadIdx.x;
    if (i >= NM) return;
    float v = acc[i] + bias[i % M];
    if (relu) v = fmaxf(v, 0.f);
    if (outf) outf[i] = v;
    if (outb) outb[i] = (__bf16)v;
}

// ---------------- log_softmax over 32 classes: one wave32 per node ----------------

__global__ void k_log_softmax32(const float* __restrict__ h, float* __restrict__ out, int N) {
    int w    = (blockIdx.x * blockDim.x + threadIdx.x) >> 5;
    int lane = threadIdx.x & 31;
    if (w >= N) return;
    float v = h[(long)w * 32 + lane];
    float m = v;
#pragma unroll
    for (int off = 16; off > 0; off >>= 1) m = fmaxf(m, __shfl_xor(m, off, 32));
    float ex = __expf(v - m);
    float sm = ex;
#pragma unroll
    for (int off = 16; off > 0; off >>= 1) sm += __shfl_xor(sm, off, 32);
    out[(long)w * 32 + lane] = v - m - __logf(sm);
}

// ---------------- host driver ----------------

static inline long cdivl(long a, long b) { return (a + b - 1) / b; }

extern "C" void kernel_launch(void* const* d_in, const int* in_sizes, int n_in,
                              void* d_out, int out_size, void* d_ws, size_t ws_size,
                              hipStream_t stream) {
    (void)n_in; (void)out_size; (void)ws_size;

    const int IN = 256, HID = 64, HEADS = 8, OUT = 32;
    const int M12 = HEADS * HID;                 // 512
    const int N   = in_sizes[0] / IN;            // 10000
    const int E   = in_sizes[13] / 2;            // 320000
    const int Etot = E + N;                      // + self loops

    const float* x   = (const float*)d_in[0];
    const float* W1  = (const float*)d_in[1];
    const float* a1s = (const float*)d_in[2];
    const float* a1d = (const float*)d_in[3];
    const float* b1  = (const float*)d_in[4];
    const float* W2  = (const float*)d_in[5];
    const float* a2s = (const float*)d_in[6];
    const float* a2d = (const float*)d_in[7];
    const float* b2  = (const float*)d_in[8];
    const float* W3  = (const float*)d_in[9];
    const float* a3s = (const float*)d_in[10];
    const float* a3d = (const float*)d_in[11];
    const float* b3  = (const float*)d_in[12];
    const int*   ei  = (const int*)d_in[13];
    const int* esrc = ei;
    const int* edst = ei + E;
    float* outp = (float*)d_out;

    // bump allocator on workspace
    char* ws = (char*)d_ws;
    auto alloc = [&](size_t bytes) -> void* {
        void* p = (void*)ws;
        ws += (bytes + 255) & ~size_t(255);
        return p;
    };
    __bf16* hb   = (__bf16*)alloc((size_t)N * M12 * sizeof(__bf16));   // layer input, bf16
    __bf16* Wt   = (__bf16*)alloc((size_t)M12 * M12 * sizeof(__bf16)); // transposed weights
    float* h     = (float*)alloc((size_t)N * M12 * sizeof(float));     // GEMM out
    float* acc   = (float*)alloc((size_t)N * M12 * sizeof(float));     // aggregation accum
    float* e_src = (float*)alloc((size_t)N * HEADS * sizeof(float));
    float* e_dst = (float*)alloc((size_t)N * HEADS * sizeof(float));
    float* emax  = (float*)alloc((size_t)N * HEADS * sizeof(float));
    float* denom = (float*)alloc((size_t)N * HEADS * sizeof(float));
    float* h3f   = (float*)alloc((size_t)N * OUT * sizeof(float));     // final pre-softmax

    const int B = 256;

    auto run_layer = [&](const float* W, const float* as, const float* ad,
                         const float* bias, int K, int M, int H, int F, int logF,
                         int relu, float* outf, __bf16* outb) {
        // weights -> bf16 transposed [M,K]
        k_transpose_bf16<<<cdivl((long)K * M, B), B, 0, stream>>>(W, Wt, K, M);
        // h = hb @ W  (WMMA); 16x64 tile/wave when M%64==0, else 16x32
        if (M % 64 == 0) {
            long waves = (long)(N / 16) * (M / 64);
            k_wmma_gemm_bf16<4><<<cdivl(waves * 32, 128), 128, 0, stream>>>(hb, Wt, h, N, K, M);
        } else {
            long waves = (long)(N / 16) * (M / 32);
            k_wmma_gemm_bf16<2><<<cdivl(waves * 32, 128), 128, 0, stream>>>(hb, Wt, h, N, K, M);
        }
        // per-node attention scores + softmax-state init
        k_scores_init<<<cdivl((long)N * H, B), B, 0, stream>>>(
            h, as, ad, e_src, e_dst, emax, denom, N, H, F, M);
        k_fill_f32<<<cdivl((long)N * M, B), B, 0, stream>>>(acc, 0.f, (long)N * M);
        // segment softmax (3 edge passes)
        k_edge_max<<<cdivl((long)Etot * H, B), B, 0, stream>>>(
            esrc, edst, E, Etot, e_src, e_dst, emax, H);
        k_edge_expsum<<<cdivl((long)Etot * H, B), B, 0, stream>>>(
            esrc, edst, E, Etot, e_src, e_dst, emax, denom, H);
        int aggB = (M >= 256) ? 256 : 64;
        k_edge_aggregate<<<Etot, aggB, 0, stream>>>(
            esrc, edst, E, Etot, h, e_src, e_dst, emax, denom, acc, H, logF, M);
        // bias (+ReLU) and conversion for next layer
        k_epilogue<<<cdivl((long)N * M, B), B, 0, stream>>>(
            acc, bias, outf, outb, (long)N * M, M, relu);
    };

    // x -> bf16
    k_cast_bf16<<<cdivl((long)N * IN, B), B, 0, stream>>>(x, hb, (long)N * IN);

    // layer 1: 256 -> 512 (8 heads x 64), ReLU, bf16 out feeds layer 2
    run_layer(W1, a1s, a1d, b1, IN,  M12, HEADS, HID, 6, 1, nullptr, hb);
    // layer 2: 512 -> 512, ReLU
    run_layer(W2, a2s, a2d, b2, M12, M12, HEADS, HID, 6, 1, nullptr, hb);
    // layer 3: 512 -> 32 (1 head), no ReLU, f32 out
    run_layer(W3, a3s, a3d, b3, M12, OUT, 1, OUT, 5, 0, h3f, nullptr);

    // log_softmax over 32 classes, one wave per node
    k_log_softmax32<<<cdivl((long)N * 32, 128), 128, 0, stream>>>(h3f, outp, N);
}
